// DynamicAttentionRNNDecodeLayer_17265768530075
// MI455X (gfx1250) — matrix operand order repaired and checked by reading, not compiled, not run on hardware
//
#include <hip/hip_runtime.h>
#include <hip/hip_bf16.h>

#define Bn   64
#define Tn   1024
#define DIN  256
#define DMEM 512
#define Hn   256
#define KL   768   // DIN + H + H   (LSTM input)
#define NL   1024  // 4*H           (gates)
#define KA   768   // H + DMEM      (attention layer input)

typedef __attribute__((ext_vector_type(16))) __bf16 v16bf;
typedef __attribute__((ext_vector_type(8)))  float  v8f;

union FragAB { v16bf v; unsigned d[8]; unsigned short s[16]; };

__device__ inline unsigned short f2bf(float f) {
  unsigned u = __float_as_uint(f);
  u += 0x7FFFu + ((u >> 16) & 1u);          // round-to-nearest-even
  return (unsigned short)(u >> 16);
}
__device__ inline float bf2f(unsigned short s) {
  return __uint_as_float(((unsigned)s) << 16);
}
__device__ inline float sigm(float x) { return 1.0f / (1.0f + __expf(-x)); }

// ---- WMMA fragment loads -------------------------------------------------
// A (16x32, bf16, row-major source, stride ld):
// lane l: m = l&15, kb = (l>>4)*8 ; VGPR v<4: K=kb+2v{+1}; v>=4: K=kb+16+2(v-4){+1}
__device__ inline v16bf load_a(const unsigned short* A, int ld, int row0, int k0, int lane) {
  FragAB f;
  const unsigned short* p = A + (size_t)(row0 + (lane & 15)) * ld + k0 + ((lane >> 4) << 3);
#pragma unroll
  for (int v = 0; v < 4; ++v) f.d[v] = *(const unsigned*)(p + 2 * v);
#pragma unroll
  for (int v = 0; v < 4; ++v) f.d[4 + v] = *(const unsigned*)(p + 16 + 2 * v);
  return f.v;
}
// B (32x16, bf16) pre-swizzled so each lane reads 16 contiguous bf16 (32B).
__device__ inline v16bf load_b(const unsigned short* swz, int blk, int lane) {
  FragAB f;
  const unsigned* q = (const unsigned*)(swz + ((size_t)blk << 9)) + lane * 8;
#pragma unroll
  for (int i = 0; i < 8; ++i) f.d[i] = q[i];
  return f.v;
}

// ---- device-wide barrier (sense via monotonically increasing generation) --
__device__ inline void grid_barrier(unsigned* bar, unsigned nwg) {
  __syncthreads();
  __threadfence();
  if (threadIdx.x == 0) {
    unsigned gen = *(volatile unsigned*)&bar[1];
    unsigned arrived = atomicAdd(&bar[0], 1u);
    if (arrived == nwg - 1u) {
      bar[0] = 0u;
      __threadfence();
      atomicAdd(&bar[1], 1u);
    } else {
      while (*(volatile unsigned*)&bar[1] == gen) __builtin_amdgcn_s_sleep(2);
    }
    __threadfence();
  }
  __syncthreads();
}

// ---- setup kernels --------------------------------------------------------
__global__ void init_state(float* h, float* c, float* attn, unsigned* bar) {
  int i = blockIdx.x * 256 + threadIdx.x;   // 64*256 = Bn*Hn
  h[i] = 0.f; c[i] = 0.f; attn[i] = 0.f;
  if (i < 16) bar[i] = 0u;
}

__global__ void conv_mem(const float* __restrict__ m, unsigned short* __restrict__ mb, int n) {
  for (int i = blockIdx.x * blockDim.x + threadIdx.x; i < n; i += gridDim.x * blockDim.x)
    mb[i] = f2bf(m[i]);
}

// Swizzle W [Ktot x Ntot] f32 row-major into per-(k32,n16)-block WMMA-B lane order (bf16).
// lane l holds column n=l&15; element j: K=(l>>4)*8+j for j<8, K=16+(l>>4)*8+(j-8) for j>=8.
__global__ void swz_w(const float* __restrict__ W, unsigned short* __restrict__ swz,
                      int Ktot, int Ntot) {
  int NB = Ntot >> 4;
  int total = (Ktot >> 5) * NB * 512;
  for (int e = blockIdx.x * blockDim.x + threadIdx.x; e < total; e += gridDim.x * blockDim.x) {
    int blk = e >> 9, idx = e & 511;
    int lane = idx >> 4, j = idx & 15;
    int kblk = blk / NB, nblk = blk - kblk * NB;
    int K = kblk * 32 + ((lane >> 4) << 3) + (j & 7) + ((j >> 3) << 4);
    int N = nblk * 16 + (lane & 15);
    swz[e] = f2bf(W[(size_t)K * Ntot + N]);
  }
}

// keys = memory @ W_mem : M=B*T=65536, K=512, N=256, bf16 in / bf16 out, f32 accum
__launch_bounds__(256)
__global__ void keys_gemm(const unsigned short* __restrict__ memb,
                          const unsigned short* __restrict__ Wm,
                          unsigned short* __restrict__ keys) {
  int lane = threadIdx.x & 31;
  int gw = blockIdx.x * 8 + (threadIdx.x >> 5);   // one 16x16 tile per wave
  int mt = gw >> 4, nt = gw & 15;
  v8f acc = {};
  for (int k = 0; k < DMEM / 32; ++k) {
    v16bf a = load_a(memb, DMEM, mt * 16, k * 32, lane);
    v16bf b = load_b(Wm, k * 16 + nt, lane);
    acc = __builtin_amdgcn_wmma_f32_16x16x32_bf16(false, a, false, b, (short)0, acc, false, false);
  }
  int row = mt * 16 + ((lane >> 4) << 3);
  int col = nt * 16 + (lane & 15);
#pragma unroll
  for (int r = 0; r < 8; ++r) keys[(size_t)(row + r) * Hn + col] = f2bf(acc[r]);
}

// ---- persistent recurrent kernel: 64 WGs x 256 thr, loops over T ----------
__launch_bounds__(256)
__global__ void rnn_persistent(
    const float* __restrict__ x, const float* __restrict__ bl,
    const int* __restrict__ slen,
    const unsigned short* __restrict__ Wl, const unsigned short* __restrict__ Wa,
    const unsigned short* __restrict__ keys, const unsigned short* __restrict__ memb,
    float* __restrict__ out,
    float* __restrict__ h, float* __restrict__ c, float* __restrict__ attn,
    float* __restrict__ hnew, unsigned short* __restrict__ zin,
    unsigned short* __restrict__ hctx, float* __restrict__ z,
    unsigned* __restrict__ bar) {
  __shared__ float s_h[Hn];
  __shared__ float s_align[Tn];
  __shared__ float s_red[256];
  const int tid  = threadIdx.x;
  const int wg   = blockIdx.x;        // == batch index for attention phases
  const int lane = tid & 31;
  const int wid  = tid >> 5;
  const int gtid = wg * 256 + tid;    // 0..16383  (batch = gtid>>8 == wg)
  const int gw   = wg * 8 + wid;      // global wave id, 0..511
  const int myj  = gtid & 255;

  for (int t = 0; t < Tn; ++t) {
    // Phase 1: zin = bf16([x_t ; attn ; h])   (64 x 768)
#pragma unroll
    for (int i = 0; i < 3; ++i) {
      int e = gtid + i * 16384;               // < 49152
      int b = e / KL, col = e - b * KL;
      float v;
      if (col < DIN)            v = x[((size_t)b * Tn + t) * DIN + col];
      else if (col < DIN + Hn)  v = attn[b * Hn + (col - DIN)];
      else                      v = h[b * Hn + (col - DIN - Hn)];
      zin[e] = f2bf(v);
    }
    grid_barrier(bar + 0, Bn);

    // Phase 2: z = zin @ W_lstm   (64x768)@(768x1024), 256 tiles / 512 waves
    if (gw < 256) {
      int mt = gw >> 6, nt = gw & 63;
      v8f acc = {};
      for (int k = 0; k < KL / 32; ++k) {
        v16bf a = load_a(zin, KL, mt * 16, k * 32, lane);
        v16bf b = load_b(Wl, k * 64 + nt, lane);
        acc = __builtin_amdgcn_wmma_f32_16x16x32_bf16(false, a, false, b, (short)0, acc, false, false);
      }
      int row = mt * 16 + ((lane >> 4) << 3);
      int col = nt * 16 + (lane & 15);
#pragma unroll
      for (int r = 0; r < 8; ++r) z[(size_t)(row + r) * NL + col] = acc[r];
    }
    grid_barrier(bar + 2, Bn);

    // Phase 3: LSTM gates; WG b owns batch b's 256 lanes
    {
      float zi = z[(size_t)wg * NL + myj]       + bl[myj];
      float zj = z[(size_t)wg * NL + 256 + myj] + bl[256 + myj];
      float zf = z[(size_t)wg * NL + 512 + myj] + bl[512 + myj];
      float zo = z[(size_t)wg * NL + 768 + myj] + bl[768 + myj];
      float cold = c[gtid];
      float cn = cold * sigm(zf + 1.0f) + sigm(zi) * tanhf(zj);
      float hn = tanhf(cn) * sigm(zo);
      bool valid = t < slen[wg];
      c[gtid] = valid ? cn : cold;
      float hold = h[gtid];
      h[gtid] = valid ? hn : hold;
      hnew[gtid] = hn;
      hctx[(size_t)wg * KA + myj] = f2bf(hn);
    }
    __syncthreads();

    // Phase 4: score = h_new . keys[b], masked softmax over T (per-WG)
    s_h[tid] = hnew[wg * Hn + tid];
    __syncthreads();
    const int sl = slen[wg];
    float sc[4];
#pragma unroll
    for (int q = 0; q < 4; ++q) {
      int ti = tid + q * 256;
      const unsigned* kp = (const unsigned*)(keys + ((size_t)wg * Tn + ti) * Hn);
      float acc = 0.f;
      for (int kk = 0; kk < Hn / 2; ++kk) {
        unsigned d = kp[kk];
        acc += s_h[2 * kk]     * __uint_as_float(d << 16);
        acc += s_h[2 * kk + 1] * __uint_as_float(d & 0xffff0000u);
      }
      sc[q] = (ti < sl) ? acc : -1e9f;
    }
    float mx = fmaxf(fmaxf(sc[0], sc[1]), fmaxf(sc[2], sc[3]));
    s_red[tid] = mx; __syncthreads();
    for (int s = 128; s > 0; s >>= 1) { if (tid < s) s_red[tid] = fmaxf(s_red[tid], s_red[tid + s]); __syncthreads(); }
    float gm = s_red[0]; __syncthreads();
    float es[4], lsum = 0.f;
#pragma unroll
    for (int q = 0; q < 4; ++q) { es[q] = __expf(sc[q] - gm); lsum += es[q]; }
    s_red[tid] = lsum; __syncthreads();
    for (int s = 128; s > 0; s >>= 1) { if (tid < s) s_red[tid] += s_red[tid + s]; __syncthreads(); }
    float inv = 1.0f / s_red[0]; __syncthreads();
#pragma unroll
    for (int q = 0; q < 4; ++q) s_align[tid + q * 256] = es[q] * inv;
    __syncthreads();

    // Phase 5: ctx = sum_t align[t] * memory[b,t,:]  (512 outputs, 2/thread)
    {
      float a0 = 0.f, a1 = 0.f;
      const unsigned short* mb = memb + (size_t)wg * Tn * DMEM;
      for (int ti = 0; ti < Tn; ++ti) {
        float al = s_align[ti];
        const unsigned short* mp = mb + (size_t)ti * DMEM;
        a0 += al * bf2f(mp[tid]);
        a1 += al * bf2f(mp[tid + 256]);
      }
      hctx[(size_t)wg * KA + Hn + tid]       = f2bf(a0);
      hctx[(size_t)wg * KA + Hn + 256 + tid] = f2bf(a1);
    }
    grid_barrier(bar + 4, Bn);

    // Phase 6: attn_new = [h_new;ctx] @ W_attn ; seq-length gating; output
    if (gw < 64) {
      int mt = gw >> 4, nt = gw & 15;
      v8f acc = {};
      for (int k = 0; k < KA / 32; ++k) {
        v16bf a = load_a(hctx, KA, mt * 16, k * 32, lane);
        v16bf b = load_b(Wa, k * 16 + nt, lane);
        acc = __builtin_amdgcn_wmma_f32_16x16x32_bf16(false, a, false, b, (short)0, acc, false, false);
      }
      int row = mt * 16 + ((lane >> 4) << 3);
      int col = nt * 16 + (lane & 15);
#pragma unroll
      for (int r = 0; r < 8; ++r) {
        int b_ = row + r;
        bool valid = t < slen[b_];
        float v = acc[r];
        out[((size_t)b_ * Tn + t) * Hn + col] = valid ? v : 0.f;
        if (valid) attn[b_ * Hn + col] = v;
      }
    }
    grid_barrier(bar + 6, Bn);
  }
}

extern "C" void kernel_launch(void* const* d_in, const int* in_sizes, int n_in,
                              void* d_out, int out_size, void* d_ws, size_t ws_size,
                              hipStream_t stream) {
  (void)in_sizes; (void)n_in; (void)out_size; (void)ws_size;
  const float* x    = (const float*)d_in[0];
  const float* mem  = (const float*)d_in[1];
  const float* Wl_f = (const float*)d_in[2];
  const float* bl   = (const float*)d_in[3];
  const float* Wm_f = (const float*)d_in[4];
  const float* Wa_f = (const float*)d_in[5];
  const int*   slen = (const int*)d_in[6];
  float* out = (float*)d_out;

  char* ws = (char*)d_ws;
  size_t o = 0;
  auto take = [&](size_t bytes) -> char* {
    char* p = ws + o; o = (o + bytes + 255) & ~(size_t)255; return p;
  };
  unsigned*       bar  = (unsigned*)take(64 * sizeof(unsigned));
  float*          h    = (float*)take((size_t)Bn * Hn * 4);
  float*          c    = (float*)take((size_t)Bn * Hn * 4);
  float*          attn = (float*)take((size_t)Bn * Hn * 4);
  float*          hnew = (float*)take((size_t)Bn * Hn * 4);
  float*          z    = (float*)take((size_t)Bn * NL * 4);
  unsigned short* zin  = (unsigned short*)take((size_t)Bn * KL * 2);
  unsigned short* hctx = (unsigned short*)take((size_t)Bn * KA * 2);
  unsigned short* Wl_s = (unsigned short*)take((size_t)KL * NL * 2);
  unsigned short* Wa_s = (unsigned short*)take((size_t)KA * Hn * 2);
  unsigned short* Wm_s = (unsigned short*)take((size_t)DMEM * Hn * 2);
  unsigned short* memb = (unsigned short*)take((size_t)Bn * Tn * DMEM * 2);
  unsigned short* keys = (unsigned short*)take((size_t)Bn * Tn * Hn * 2);

  hipLaunchKernelGGL(init_state, dim3(64), dim3(256), 0, stream, h, c, attn, bar);
  hipLaunchKernelGGL(conv_mem, dim3(2048), dim3(256), 0, stream, mem, memb, Bn * Tn * DMEM);
  hipLaunchKernelGGL(swz_w, dim3(1024), dim3(256), 0, stream, Wl_f, Wl_s, KL, NL);
  hipLaunchKernelGGL(swz_w, dim3(512),  dim3(256), 0, stream, Wa_f, Wa_s, KA, Hn);
  hipLaunchKernelGGL(swz_w, dim3(512),  dim3(256), 0, stream, Wm_f, Wm_s, DMEM, Hn);
  hipLaunchKernelGGL(keys_gemm, dim3(8192), dim3(256), 0, stream, memb, Wm_s, keys);
  hipLaunchKernelGGL(rnn_persistent, dim3(Bn), dim3(256), 0, stream,
                     x, bl, slen, Wl_s, Wa_s, keys, memb, out,
                     h, c, attn, hnew, zin, hctx, z, bar);
}
